// EditFactorizedEnergyNet_23570780520486
// MI455X (gfx1250) — matrix operand order repaired and checked by reading, not compiled, not run on hardware
//
#include <hip/hip_runtime.h>

typedef __attribute__((ext_vector_type(16))) _Float16 v16h;
typedef __attribute__((ext_vector_type(8)))  float    v8f;

#define Bn 8
#define Kn 256
#define DAc 64
#define DZc 32
#define Hn 128
#define KKc (Kn*Kn)
#define HUGE_E 1000000.0f

// ---------------------------------------------------------------------------
// Kernel 1: per-agent layer-1 partial sums (factorized first layer).
//   Ai[b,k,:] = b1 + ai.W1[Ai-cols] - (linear geom i-part) + zi_m/zi_lv parts
//   Bj[b,k,:] =      aj.W1[Aj-cols] + (linear geom j-part) + zj_m/zj_lv parts
// ---------------------------------------------------------------------------
__device__ __forceinline__ void row_pre(const float* __restrict__ a,
                                        const float* __restrict__ zmr,
                                        const float* __restrict__ zlr,
                                        const float* __restrict__ W1,
                                        const float* __restrict__ b1,
                                        int h, int offAi, int offAj, int offG,
                                        float& ai_out, float& bj_out) {
    float ai = b1[h], bj = 0.f;
    for (int f = 0; f < DAc; ++f) {
        float av = a[f];
        ai = fmaf(av, W1[(offAi + f) * Hn + h], ai);
        bj = fmaf(av, W1[(offAj + f) * Hn + h], bj);
    }
    float g0 = W1[(offG + 0) * Hn + h], g1 = W1[(offG + 1) * Hn + h];
    float g2 = W1[(offG + 2) * Hn + h], g3 = W1[(offG + 3) * Hn + h];
    float lin = a[0] * g0 + a[1] * g1 + a[3] * g2 + a[4] * g3;
    ai -= lin; bj += lin;
    for (int f = 0; f < DZc; ++f) {
        ai = fmaf(zmr[f], W1[(133 + f) * Hn + h], ai);
        ai = fmaf(zlr[f], W1[(165 + f) * Hn + h], ai);
        bj = fmaf(zmr[f], W1[(197 + f) * Hn + h], bj);
        bj = fmaf(zlr[f], W1[(229 + f) * Hn + h], bj);
    }
    ai_out = ai; bj_out = bj;
}

__global__ __launch_bounds__(128) void fen_pre_kernel(
    const float* __restrict__ af, const float* __restrict__ zm,
    const float* __restrict__ zlv,
    const float* __restrict__ W1d, const float* __restrict__ b1d,
    const float* __restrict__ W1n, const float* __restrict__ b1n,
    float* __restrict__ AiD, float* __restrict__ BjD,
    float* __restrict__ AiN, float* __restrict__ BjN) {
    int row = blockIdx.x;            // b*K + k
    int h = threadIdx.x;             // hidden index
    const float* a   = af  + (size_t)row * DAc;
    const float* zmr = zm  + (size_t)row * DZc;
    const float* zlr = zlv + (size_t)row * DZc;
    float aiD, bjD, aiN, bjN;
    row_pre(a, zmr, zlr, W1d, b1d, h, 0, 64, 128, aiD, bjD);   // dir: ai@0, aj@64, geom@128
    row_pre(a, zmr, zlr, W1n, b1n, h, 5, 69, 0,   aiN, bjN);   // none: geom@0, ai@5, aj@69
    size_t o = (size_t)row * Hn + h;
    AiD[o] = aiD; BjD[o] = bjD; AiN[o] = aiN; BjN[o] = bjN;
}

// ---------------------------------------------------------------------------
// Kernel 2: W2 (128x128 f32, k-major) -> W2T (n-major rows of 128, f16)
// ---------------------------------------------------------------------------
__global__ __launch_bounds__(256) void fen_packw2_kernel(
    const float* __restrict__ W2d, const float* __restrict__ W2n,
    _Float16* __restrict__ W2T) {
    int idx = blockIdx.x * 256 + threadIdx.x;   // 0 .. 2*128*128-1
    int mlp = idx >> 14;
    int r = idx & 16383;
    int n = r >> 7, k = r & 127;
    const float* W2 = mlp ? W2n : W2d;
    W2T[idx] = (_Float16)W2[k * Hn + n];
}

// ---------------------------------------------------------------------------
// Kernel 3: main pair-MLP kernel. One block = (b, i, jhalf); 8 waves, each a
// 16-j tile. Layer 2 via v_wmma_f32_16x16x32_f16; layer 3 fused on fragments.
// ---------------------------------------------------------------------------
__global__ __launch_bounds__(256) void fen_pair_kernel(
    const float* __restrict__ AiD, const float* __restrict__ BjD,
    const float* __restrict__ AiN, const float* __restrict__ BjN,
    const _Float16* __restrict__ gW2T,
    const float* __restrict__ af, const float* __restrict__ amask,
    const float* __restrict__ W1d, const float* __restrict__ W1n,
    const float* __restrict__ b2d, const float* __restrict__ b2n,
    const float* __restrict__ w3d, const float* __restrict__ b3d,
    const float* __restrict__ w3n, const float* __restrict__ b3n,
    float* __restrict__ eDir, float* __restrict__ rawNone) {
    __shared__ _Float16 sW2T[Hn * Hn];        // 32 KB, reloaded per MLP
    __shared__ _Float16 sH1[8 * 16 * Hn];     // 32 KB, 4 KB per wave

    int tid = threadIdx.x;
    int w = tid >> 5, lane = tid & 31;
    int hi = lane >> 4, ln = lane & 15;
    int bid = blockIdx.x;
    int b = bid / (Kn * 2);
    int rem = bid % (Kn * 2);
    int i = rem >> 1;
    int jhalf = rem & 1;
    int j0 = jhalf * 128 + w * 16;
    int rowi = b * Kn + i;
    float xi0 = af[(size_t)rowi * DAc + 0];
    float xi1 = af[(size_t)rowi * DAc + 1];
    float mi = amask[rowi];
    _Float16* myH1 = sH1 + w * (16 * Hn);

    for (int mlp = 0; mlp < 2; ++mlp) {
        __syncthreads();  // previous readers of sW2T done
        const _Float16* src = gW2T + mlp * (Hn * Hn);
        for (int t = tid; t < (Hn * Hn) / 8; t += 256)
            ((uint4*)sW2T)[t] = ((const uint4*)src)[t];
        __syncthreads();

        const float* Ai    = mlp ? AiN : AiD;
        const float* Bj    = mlp ? BjN : BjD;
        const float* wdrow = mlp ? (W1n + 4 * Hn) : (W1d + 132 * Hn);  // dist column of W1
        const float* b2    = mlp ? b2n : b2d;
        const float* w3    = mlp ? w3n : w3d;
        float b3v          = mlp ? b3n[0] : b3d[0];
        float* target      = mlp ? rawNone : eDir;

        // ----- build h1 tile (16 pairs x 128) in LDS, f16, relu applied -----
        float4 aiv = *(const float4*)(Ai + (size_t)rowi * Hn + 4 * lane);
        float4 wdv = *(const float4*)(wdrow + 4 * lane);
        for (int r = 0; r < 16; ++r) {
            int rowj = b * Kn + j0 + r;
            float4 bjv = *(const float4*)(Bj + (size_t)rowj * Hn + 4 * lane);
            float dx = af[(size_t)rowj * DAc + 0] - xi0;
            float dy = af[(size_t)rowj * DAc + 1] - xi1;
            float dist = sqrtf(dx * dx + dy * dy);
            union { _Float16 h[4]; uint2 u; } pk;
            pk.h[0] = (_Float16)fmaxf(fmaf(dist, wdv.x, aiv.x + bjv.x), 0.f);
            pk.h[1] = (_Float16)fmaxf(fmaf(dist, wdv.y, aiv.y + bjv.y), 0.f);
            pk.h[2] = (_Float16)fmaxf(fmaf(dist, wdv.z, aiv.z + bjv.z), 0.f);
            pk.h[3] = (_Float16)fmaxf(fmaf(dist, wdv.w, aiv.w + bjv.w), 0.f);
            *(uint2*)(myH1 + r * Hn + 4 * lane) = pk.u;
        }

        // ----- A fragments: 16-bit A 16x32 layout -----
        // lane (hi,ln): row M=ln; halves t0..7 = K(s*32+8*hi ..), t8..15 = K(s*32+16+8*hi ..)
        v16h afr[4];
        #pragma unroll
        for (int s = 0; s < 4; ++s) {
            const _Float16* ap = myH1 + ln * Hn + s * 32 + 8 * hi;
            ((uint4*)&afr[s])[0] = *(const uint4*)(ap);
            ((uint4*)&afr[s])[1] = *(const uint4*)(ap + 16);
        }

        // ----- layer 2 (WMMA) + fused relu + layer-3 dot -----
        v8f part = {0.f, 0.f, 0.f, 0.f, 0.f, 0.f, 0.f, 0.f};
        #pragma unroll
        for (int c = 0; c < 8; ++c) {
            float b2v = b2[c * 16 + ln];
            v8f acc = {b2v, b2v, b2v, b2v, b2v, b2v, b2v, b2v};
            #pragma unroll
            for (int s = 0; s < 4; ++s) {
                // B fragment: col N = c*16+ln, K = s*32+8*hi .. +7, then s*32+16+8*hi .. +7
                v16h bfr;
                const _Float16* bp = sW2T + (c * 16 + ln) * Hn + s * 32 + 8 * hi;
                ((uint4*)&bfr)[0] = *(const uint4*)(bp);
                ((uint4*)&bfr)[1] = *(const uint4*)(bp + 16);
                acc = __builtin_amdgcn_wmma_f32_16x16x32_f16(
                    false, afr[s], false, bfr, (short)0, acc, false, false);
            }
            float w3v = w3[c * 16 + ln];
            #pragma unroll
            for (int v = 0; v < 8; ++v)
                part[v] += fmaxf(acc[v], 0.f) * w3v;
        }

        // ----- reduce N across the 16 lanes of each half; store -----
        #pragma unroll
        for (int v = 0; v < 8; ++v) {
            float p = part[v];
            p += __shfl_xor(p, 1, 32);
            p += __shfl_xor(p, 2, 32);
            p += __shfl_xor(p, 4, 32);
            p += __shfl_xor(p, 8, 32);
            if (ln == 0) {   // lanes 0 (rows 0..7) and 16 (rows 8..15)
                int m = v + 8 * hi;
                int j = j0 + m;
                float mj = amask[b * Kn + j];
                bool keep = (mi * mj > 0.5f) && (i != j);
                target[(size_t)b * KKc + (size_t)i * Kn + j] = keep ? (p + b3v) : HUGE_E;
            }
        }
    }
}

// ---------------------------------------------------------------------------
// Kernel 4: symmetrize e_none = 0.5*(raw + raw^T)
// ---------------------------------------------------------------------------
__global__ __launch_bounds__(256) void fen_sym_kernel(
    const float* __restrict__ raw, float* __restrict__ outN) {
    int idx = blockIdx.x * 256 + threadIdx.x;  // 0 .. B*K*K-1
    int b = idx / KKc;
    int r = idx % KKc;
    int i = r / Kn, j = r % Kn;
    outN[idx] = 0.5f * (raw[(size_t)b * KKc + (size_t)i * Kn + j] +
                        raw[(size_t)b * KKc + (size_t)j * Kn + i]);
}

extern "C" void kernel_launch(void* const* d_in, const int* in_sizes, int n_in,
                              void* d_out, int out_size, void* d_ws, size_t ws_size,
                              hipStream_t stream) {
    (void)in_sizes; (void)n_in; (void)out_size; (void)ws_size;
    const float* af    = (const float*)d_in[0];
    const float* zm    = (const float*)d_in[1];
    const float* zlv   = (const float*)d_in[2];
    const float* amask = (const float*)d_in[3];
    const float* W1d = (const float*)d_in[4];  const float* b1d = (const float*)d_in[5];
    const float* W2d = (const float*)d_in[6];  const float* b2d = (const float*)d_in[7];
    const float* W3d = (const float*)d_in[8];  const float* b3d = (const float*)d_in[9];
    const float* W1n = (const float*)d_in[10]; const float* b1n = (const float*)d_in[11];
    const float* W2n = (const float*)d_in[12]; const float* b2n = (const float*)d_in[13];
    const float* W3n = (const float*)d_in[14]; const float* b3n = (const float*)d_in[15];

    float* out = (float*)d_out;                 // [e_dir | e_none], each B*K*K f32
    float* wsf = (float*)d_ws;
    const size_t NRH = (size_t)Bn * Kn * Hn;    // 262144
    float* AiD = wsf;
    float* BjD = wsf + NRH;
    float* AiN = wsf + 2 * NRH;
    float* BjN = wsf + 3 * NRH;
    float* raw = wsf + 4 * NRH;                 // B*K*K f32 (masked e_none, pre-symmetrize)
    _Float16* W2T = (_Float16*)(wsf + 4 * NRH + (size_t)Bn * KKc);

    fen_pre_kernel<<<Bn * Kn, 128, 0, stream>>>(af, zm, zlv, W1d, b1d, W1n, b1n,
                                                AiD, BjD, AiN, BjN);
    fen_packw2_kernel<<<(2 * Hn * Hn) / 256, 256, 0, stream>>>(W2d, W2n, W2T);
    fen_pair_kernel<<<Bn * Kn * 2, 256, 0, stream>>>(
        AiD, BjD, AiN, BjN, W2T, af, amask, W1d, W1n,
        b2d, b2n, W3d, b3d, W3n, b3n, out, raw);
    fen_sym_kernel<<<((size_t)Bn * KKc) / 256, 256, 0, stream>>>(raw, out + (size_t)Bn * KKc);
}
